// SVDHead_matching2_76544907149614
// MI455X (gfx1250) — matrix operand order repaired and checked by reading, not compiled, not run on hardware
//
#include <hip/hip_runtime.h>
#include <hip/hip_bf16.h>
#include <math.h>

typedef __attribute__((ext_vector_type(16))) _Float16 v16h;
typedef __attribute__((ext_vector_type(8)))  float    v8f;

#define B_   16
#define DK_  128
#define N_   2048
#define NS_  15
#define RB_  2      // 16-row blocks per wave (32 rows)
#define CT_  16     // 16-wide column tiles per wave (256 cols)
#define NT_  (N_ / 16)   // 128 tiles per matrix edge

// fragment record: 32 lanes x 32 bytes (v16h), index ((b*NT_+tile)*4+kc)*32+lane
__device__ __forceinline__ size_t fragIdx(int b, int tile, int kc, int lane) {
  return ((((size_t)b * NT_ + tile) * 4 + kc) * 32) + lane;
}

// ---------------------------------------------------------------------------
// Staging kernels: gather strided f32 embeddings ONCE, convert to f16, and
// store fragments pre-swizzled into the WMMA register layout (ISA 7.12.2).
// One wave per (tile, kc) fragment.
// ---------------------------------------------------------------------------
__global__ __launch_bounds__(32)
void stage_a_frags(const float* __restrict__ A, v16h* __restrict__ Astage)
{
  const int lane = threadIdx.x;
  const int half = lane >> 4;
  const int l15  = lane & 15;
  const int nt   = blockIdx.x;   // row tile
  const int kc   = blockIdx.y;   // K chunk of 32
  const int b    = blockIdx.z;

  const float* Ab = A + (size_t)b * DK_ * N_;   // A[d][n]
  const int n = nt * 16 + l15;

  // A 16x32 f16 layout: lanes 0-15 M=0..15, V0..3->K=0..7, V4..7->K=16..23;
  // lanes 16-31: K offsets +8.
  v16h af;
  #pragma unroll
  for (int v = 0; v < 8; ++v) {
    const int kin = (v < 4) ? (half * 8 + 2 * v) : (16 + half * 8 + 2 * (v - 4));
    const int kg  = kc * 32 + kin;
    af[2 * v]     = (_Float16)Ab[(size_t)kg       * N_ + n];
    af[2 * v + 1] = (_Float16)Ab[(size_t)(kg + 1) * N_ + n];
  }
  Astage[fragIdx(b, nt, kc, lane)] = af;
}

__global__ __launch_bounds__(32)
void stage_b_frags(const float* __restrict__ Bm, v16h* __restrict__ Bstage)
{
  const int lane = threadIdx.x;
  const int half = lane >> 4;
  const int l15  = lane & 15;
  const int mt   = blockIdx.x;   // col tile
  const int kc   = blockIdx.y;
  const int b    = blockIdx.z;

  const float* Bb = Bm + (size_t)b * DK_ * N_;  // B[d][m]
  const int m = mt * 16 + l15;

  // B 32x16 f16 layout: lane = N, V0..3->K=0..7, V4..7->K=16..23; lanes 16-31: K offsets +8.
  v16h bf;
  #pragma unroll
  for (int e = 0; e < 16; ++e)
    bf[e] = (_Float16)Bb[(size_t)(kc * 32 + ((e < 8) ? (half * 8 + e) : (16 + half * 8 + e - 8))) * N_ + m];
  Bstage[fragIdx(b, mt, kc, lane)] = bf;
}

// ---------------------------------------------------------------------------
// Kernel 1: scores[b] = (srcE[b]^T @ tgtE[b]) / sqrt(128)  via v_wmma f16.
// One wave per block: 32 rows x 256 cols. Fragments load as coalesced b128s.
// ---------------------------------------------------------------------------
__global__ __launch_bounds__(32)
void gemm_scores_wmma(const v16h* __restrict__ Astage,
                      const v16h* __restrict__ Bstage,
                      float* __restrict__ S)
{
  const int lane = threadIdx.x;
  const int half = lane >> 4;
  const int l15  = lane & 15;
  const int b    = blockIdx.z;
  const int nt0  = blockIdx.x * RB_;           // first row tile
  const int mt0  = blockIdx.y * CT_;           // first col tile

  float* Sb = S + (size_t)b * N_ * N_;

  v16h af[RB_][4];
  #pragma unroll
  for (int rb = 0; rb < RB_; ++rb)
    #pragma unroll
    for (int kc = 0; kc < 4; ++kc)
      af[rb][kc] = Astage[fragIdx(b, nt0 + rb, kc, lane)];

  const float scale = 0.088388347648318447f;  // 1/sqrt(128)

  for (int ct = 0; ct < CT_; ++ct) {
    const int mt = mt0 + ct;
    if (ct + 1 < CT_)
      __builtin_prefetch(&Bstage[fragIdx(b, mt + 1, 0, lane)], 0, 1);

    v8f acc[RB_];
    #pragma unroll
    for (int rb = 0; rb < RB_; ++rb) acc[rb] = (v8f){0,0,0,0,0,0,0,0};

    #pragma unroll
    for (int kc = 0; kc < 4; ++kc) {
      const v16h bf = Bstage[fragIdx(b, mt, kc, lane)];
      #pragma unroll
      for (int rb = 0; rb < RB_; ++rb)
        acc[rb] = __builtin_amdgcn_wmma_f32_16x16x32_f16(
            false, af[rb][kc], false, bf, (short)0, acc[rb], false, false);
    }

    // C f32 layout: VGPR r -> rows (half? r+8 : r), N = l15
    #pragma unroll
    for (int rb = 0; rb < RB_; ++rb) {
      const int rowbase = (nt0 + rb) * 16 + half * 8;
      #pragma unroll
      for (int r = 0; r < 8; ++r)
        Sb[(size_t)(rowbase + r) * N_ + mt * 16 + l15] = acc[rb][r] * scale;
    }
  }
}

// ---------------------------------------------------------------------------
// Kernel 2: in-place row softmax; record row top value (=1/rowsum) + argmax
// ---------------------------------------------------------------------------
__global__ __launch_bounds__(256)
void softmax_rows(float* __restrict__ S, float* __restrict__ rowVal,
                  int* __restrict__ rowIdx)
{
  const int row = blockIdx.x;
  const int tid = threadIdx.x;
  float* p = S + (size_t)row * N_;

  __shared__ float sv[256];
  __shared__ int   si[256];

  float x[8];
  float mx = -__builtin_inff();
  int   mi = 0;
  #pragma unroll
  for (int k = 0; k < 8; ++k) {
    const int m = tid + k * 256;
    x[k] = p[m];
    if (x[k] > mx) { mx = x[k]; mi = m; }
  }
  sv[tid] = mx; si[tid] = mi;
  __syncthreads();
  for (int s = 128; s > 0; s >>= 1) {
    if (tid < s) {
      const float v2 = sv[tid + s]; const int i2 = si[tid + s];
      if (v2 > sv[tid] || (v2 == sv[tid] && i2 < si[tid])) { sv[tid] = v2; si[tid] = i2; }
    }
    __syncthreads();
  }
  const float rmax = sv[0];
  const int   ridx = si[0];
  __syncthreads();

  float e[8], loc = 0.f;
  #pragma unroll
  for (int k = 0; k < 8; ++k) { e[k] = expf(x[k] - rmax); loc += e[k]; }
  sv[tid] = loc;
  __syncthreads();
  for (int s = 128; s > 0; s >>= 1) {
    if (tid < s) sv[tid] += sv[tid + s];
    __syncthreads();
  }
  const float inv = 1.f / sv[0];
  #pragma unroll
  for (int k = 0; k < 8; ++k) p[tid + k * 256] = e[k] * inv;
  if (tid == 0) { rowVal[row] = inv; rowIdx[row] = ridx; }
}

// ---------------------------------------------------------------------------
// Kernel 3: column sums of softmax matrix (for src_corr.mean)
// ---------------------------------------------------------------------------
__global__ __launch_bounds__(256)
void col_sums(const float* __restrict__ S, float* __restrict__ colsum)
{
  const int b = blockIdx.y;
  const int m = blockIdx.x * 256 + threadIdx.x;
  const float* p = S + (size_t)b * N_ * N_ + m;
  float s = 0.f;
  for (int n = 0; n < N_; ++n) s += p[(size_t)n * N_];
  colsum[b * N_ + m] = s;
}

// ---------------------------------------------------------------------------
// Kernel 4: greedy matching; champions in LDS + lazy rescan.
// ---------------------------------------------------------------------------
__global__ __launch_bounds__(256)
void greedy_match(const float* __restrict__ S, const float* __restrict__ rowValG,
                  const int* __restrict__ rowIdxG, int* __restrict__ matches)
{
  const int b = blockIdx.x, tid = threadIdx.x;
  __shared__ float    rv[N_];
  __shared__ int      ri[N_];
  __shared__ unsigned removed[N_ / 32];
  __shared__ float    redV[256];
  __shared__ int      redI[256];

  for (int n = tid; n < N_; n += 256) { rv[n] = rowValG[b * N_ + n]; ri[n] = rowIdxG[b * N_ + n]; }
  for (int w = tid; w < N_ / 32; w += 256) removed[w] = 0u;
  __syncthreads();

  const float* Sb = S + (size_t)b * N_ * N_;
  const float NEGINF = -__builtin_inff();

  for (int step = 0; step < NS_; ++step) {
    float best = NEGINF; int bi = N_;
    for (int n = tid; n < N_; n += 256) {
      const float v = rv[n];
      if (v > best || (v == best && n < bi)) { best = v; bi = n; }
    }
    redV[tid] = best; redI[tid] = bi;
    __syncthreads();
    for (int s = 128; s > 0; s >>= 1) {
      if (tid < s) {
        const float v2 = redV[tid + s]; const int i2 = redI[tid + s];
        if (v2 > redV[tid] || (v2 == redV[tid] && i2 < redI[tid])) { redV[tid] = v2; redI[tid] = i2; }
      }
      __syncthreads();
    }
    const int r = redI[0];
    const int c = ri[r];
    __syncthreads();
    if (tid == 0) {
      matches[(b * NS_ + step) * 2]     = r;
      matches[(b * NS_ + step) * 2 + 1] = c;
      rv[r] = NEGINF;
      removed[c >> 5] |= (1u << (c & 31));
    }
    __syncthreads();
    for (int n = tid; n < N_; n += 256) {
      if (rv[n] != NEGINF && ri[n] == c) {
        const float* prow = Sb + (size_t)n * N_;
        float nb = NEGINF; int nbi = 0;
        for (int m = 0; m < N_; ++m) {
          if ((removed[m >> 5] >> (m & 31)) & 1u) continue;
          const float v = prow[m];
          if (v > nb) { nb = v; nbi = m; }
        }
        rv[n] = nb; ri[n] = nbi;
      }
    }
    __syncthreads();
  }
}

// ---------------------------------------------------------------------------
// Kernel 5: means + Kabsch (Jacobi eigensolve of H^T H, right-handed frames).
// ---------------------------------------------------------------------------
__global__ __launch_bounds__(64)
void finalize_rt(const float* __restrict__ src, const float* __restrict__ tgt,
                 const float* __restrict__ colsum, const int* __restrict__ matches,
                 float* __restrict__ out)
{
  const int b = blockIdx.x, tid = threadIdx.x;
  __shared__ float red[6][64];

  const float* sp = src + (size_t)b * N_ * 3;
  const float* tp = tgt + (size_t)b * N_ * 3;
  const float* cs = colsum + b * N_;

  float a0 = 0, a1 = 0, a2 = 0, c0 = 0, c1 = 0, c2 = 0;
  for (int n = tid; n < N_; n += 64) {
    a0 += sp[3 * n]; a1 += sp[3 * n + 1]; a2 += sp[3 * n + 2];
    const float w = cs[n];
    c0 += tp[3 * n] * w; c1 += tp[3 * n + 1] * w; c2 += tp[3 * n + 2] * w;
  }
  red[0][tid] = a0; red[1][tid] = a1; red[2][tid] = a2;
  red[3][tid] = c0; red[4][tid] = c1; red[5][tid] = c2;
  __syncthreads();
  for (int s = 32; s > 0; s >>= 1) {
    if (tid < s)
      for (int j = 0; j < 6; ++j) red[j][tid] += red[j][tid + s];
    __syncthreads();
  }

  if (tid != 0) return;

  const float invN = 1.f / (float)N_;
  const float smean[3] = { red[0][0] * invN, red[1][0] * invN, red[2][0] * invN };
  const float tcorr[3] = { red[3][0] * invN, red[4][0] * invN, red[5][0] * invN };

  float spt[NS_][3], tpt[NS_][3], sm[3] = {0,0,0}, tm[3] = {0,0,0};
  for (int k = 0; k < NS_; ++k) {
    const int r = matches[(b * NS_ + k) * 2];
    const int c = matches[(b * NS_ + k) * 2 + 1];
    for (int d = 0; d < 3; ++d) {
      spt[k][d] = sp[3 * r + d]; tpt[k][d] = tp[3 * c + d];
      sm[d] += spt[k][d];        tm[d] += tpt[k][d];
    }
  }
  for (int d = 0; d < 3; ++d) { sm[d] /= (float)NS_; tm[d] /= (float)NS_; }

  float H[3][3] = {{0}};
  for (int k = 0; k < NS_; ++k)
    for (int i = 0; i < 3; ++i)
      for (int j = 0; j < 3; ++j)
        H[i][j] += (spt[k][i] - sm[i]) * (tpt[k][j] - tm[j]);

  float A[3][3], V[3][3] = {{1,0,0},{0,1,0},{0,0,1}};
  for (int i = 0; i < 3; ++i)
    for (int j = 0; j < 3; ++j) {
      float s = 0.f;
      for (int k = 0; k < 3; ++k) s += H[k][i] * H[k][j];
      A[i][j] = s;
    }
  const int PQ[3][2] = {{0,1},{0,2},{1,2}};
  for (int sweep = 0; sweep < 24; ++sweep) {
    for (int rot = 0; rot < 3; ++rot) {
      const int p = PQ[rot][0], q = PQ[rot][1];
      const float apq = A[p][q];
      if (fabsf(apq) < 1e-20f) continue;
      const float theta = (A[q][q] - A[p][p]) / (2.f * apq);
      const float t = copysignf(1.f, theta) / (fabsf(theta) + sqrtf(theta * theta + 1.f));
      const float cth = 1.f / sqrtf(t * t + 1.f);
      const float sth = t * cth;
      const float app = A[p][p], aqq = A[q][q];
      A[p][p] = app - t * apq;
      A[q][q] = aqq + t * apq;
      A[p][q] = A[q][p] = 0.f;
      const int r = 3 - p - q;
      const float arp = A[r][p], arq = A[r][q];
      A[r][p] = A[p][r] = cth * arp - sth * arq;
      A[r][q] = A[q][r] = sth * arp + cth * arq;
      for (int i = 0; i < 3; ++i) {
        const float vip = V[i][p], viq = V[i][q];
        V[i][p] = cth * vip - sth * viq;
        V[i][q] = sth * vip + cth * viq;
      }
    }
  }
  float ev[3] = { A[0][0], A[1][1], A[2][2] };
  int ord[3] = {0, 1, 2};
  if (ev[ord[0]] < ev[ord[1]]) { int t0 = ord[0]; ord[0] = ord[1]; ord[1] = t0; }
  if (ev[ord[1]] < ev[ord[2]]) { int t0 = ord[1]; ord[1] = ord[2]; ord[2] = t0; }
  if (ev[ord[0]] < ev[ord[1]]) { int t0 = ord[0]; ord[0] = ord[1]; ord[1] = t0; }

  float v0[3], v1[3];
  for (int i = 0; i < 3; ++i) { v0[i] = V[i][ord[0]]; v1[i] = V[i][ord[1]]; }

  float hv0[3], hv1[3];
  for (int i = 0; i < 3; ++i) {
    hv0[i] = H[i][0] * v0[0] + H[i][1] * v0[1] + H[i][2] * v0[2];
    hv1[i] = H[i][0] * v1[0] + H[i][1] * v1[1] + H[i][2] * v1[2];
  }
  float n0 = sqrtf(hv0[0]*hv0[0] + hv0[1]*hv0[1] + hv0[2]*hv0[2]) + 1e-30f;
  float u0[3] = { hv0[0]/n0, hv0[1]/n0, hv0[2]/n0 };
  const float d01 = hv1[0]*u0[0] + hv1[1]*u0[1] + hv1[2]*u0[2];
  float w1[3] = { hv1[0]-d01*u0[0], hv1[1]-d01*u0[1], hv1[2]-d01*u0[2] };
  float n1 = sqrtf(w1[0]*w1[0] + w1[1]*w1[1] + w1[2]*w1[2]) + 1e-30f;
  float u1[3] = { w1[0]/n1, w1[1]/n1, w1[2]/n1 };
  float u2[3] = { u0[1]*u1[2]-u0[2]*u1[1], u0[2]*u1[0]-u0[0]*u1[2], u0[0]*u1[1]-u0[1]*u1[0] };
  float v2[3] = { v0[1]*v1[2]-v0[2]*v1[1], v0[2]*v1[0]-v0[0]*v1[2], v0[0]*v1[1]-v0[1]*v1[0] };

  float R[3][3];
  for (int i = 0; i < 3; ++i)
    for (int j = 0; j < 3; ++j)
      R[i][j] = v0[i]*u0[j] + v1[i]*u1[j] + v2[i]*u2[j];

  for (int i = 0; i < 3; ++i)
    for (int j = 0; j < 3; ++j)
      out[b * 9 + i * 3 + j] = R[i][j];
  for (int i = 0; i < 3; ++i)
    out[B_ * 9 + b * 3 + i] =
        -(R[i][0]*smean[0] + R[i][1]*smean[1] + R[i][2]*smean[2]) + tcorr[i];
}

// ---------------------------------------------------------------------------
extern "C" void kernel_launch(void* const* d_in, const int* in_sizes, int n_in,
                              void* d_out, int out_size, void* d_ws, size_t ws_size,
                              hipStream_t stream)
{
  const float* srcE = (const float*)d_in[0];  // [B, DK, N]
  const float* tgtE = (const float*)d_in[1];  // [B, DK, N]
  const float* src  = (const float*)d_in[2];  // [B, N, 3]
  const float* tgt  = (const float*)d_in[3];  // [B, N, 3]
  float* out = (float*)d_out;                 // R:[B,3,3] ++ t:[B,3]

  // workspace layout
  char* ws = (char*)d_ws;
  const size_t scoresBytes = (size_t)B_ * N_ * N_ * sizeof(float);        // 256 MB
  const size_t stageBytes  = (size_t)B_ * NT_ * 4 * 32 * sizeof(v16h);    // 8 MB each
  float* scores = (float*)ws;
  float* rowVal = (float*)(ws + scoresBytes);
  int*   rowIdx = (int*)  (ws + scoresBytes + (size_t)B_ * N_ * 4);
  float* colsum = (float*)(ws + scoresBytes + (size_t)B_ * N_ * 8);
  int*   match  = (int*)  (ws + scoresBytes + (size_t)B_ * N_ * 12);
  v16h*  Astage = (v16h*) (ws + scoresBytes + (size_t)B_ * N_ * 16);
  v16h*  Bstage = (v16h*) (ws + scoresBytes + (size_t)B_ * N_ * 16 + stageBytes);

  dim3 gStage(NT_, 4, B_);                              // 128 x 4 x 16 waves
  stage_a_frags<<<gStage, 32, 0, stream>>>(srcE, Astage);
  stage_b_frags<<<gStage, 32, 0, stream>>>(tgtE, Bstage);

  dim3 gGemm(NT_ / RB_, NT_ / CT_, B_);                 // 64 x 8 x 16 waves
  gemm_scores_wmma<<<gGemm, 32, 0, stream>>>(Astage, Bstage, scores);

  softmax_rows<<<B_ * N_, 256, 0, stream>>>(scores, rowVal, rowIdx);

  col_sums<<<dim3(N_ / 256, B_), 256, 0, stream>>>(scores, colsum);

  greedy_match<<<B_, 256, 0, stream>>>(scores, rowVal, rowIdx, match);

  finalize_rt<<<B_, 64, 0, stream>>>(src, tgt, colsum, match, out);
}
